// SimplyInteraction_55637006352697
// MI455X (gfx1250) — compile-verified, matrix operand order checked by reading, not run on hardware
//
#include <hip/hip_runtime.h>

typedef __attribute__((ext_vector_type(16))) _Float16 v16h;
typedef __attribute__((ext_vector_type(8)))  _Float16 v8h;
typedef __attribute__((ext_vector_type(8)))  float    v8f;
typedef __attribute__((ext_vector_type(4)))  float    v4f;

#define XDIM 32
#define HID 64
#define INTDIM 8
#define ROWS_PER_CHUNK 128
#define ELEMS_PER_CHUNK (ROWS_PER_CHUNK * INTDIM)  // 1024
#define CHUNKS_PER_BLOCK 8
#define THREADS 256

// ---------------------------------------------------------------------------
// Pass 1: per-chunk nonzero count of the boolean mask (chunk == 128 rows)
// ---------------------------------------------------------------------------
__global__ void si_count_kernel(const int* __restrict__ mask,
                                unsigned* __restrict__ counts,
                                long long total_elems) {
    __shared__ int cnt;
    if (threadIdx.x == 0) cnt = 0;
    __syncthreads();
    long long base = (long long)blockIdx.x * ELEMS_PER_CHUNK;
    int c = 0;
#pragma unroll
    for (int i = 0; i < 4; ++i) {
        long long e = base + (long long)threadIdx.x * 4 + i;
        if (e < total_elems && mask[e] != 0) ++c;
    }
    atomicAdd(&cnt, c);
    __syncthreads();
    if (threadIdx.x == 0) counts[blockIdx.x] = (unsigned)cnt;
}

// ---------------------------------------------------------------------------
// Pass 2: single-block exclusive scan of chunk counts -> chunk base offsets
// ---------------------------------------------------------------------------
__global__ void si_scan_kernel(const unsigned* __restrict__ counts,
                               unsigned* __restrict__ offsets,
                               int nchunks) {
    __shared__ unsigned psum[THREADS];
    __shared__ unsigned pbase[THREADS];
    const int t = threadIdx.x;
    const int per = (nchunks + THREADS - 1) / THREADS;
    unsigned s = 0;
    for (int i = 0; i < per; ++i) {
        int idx = t * per + i;
        if (idx < nchunks) s += counts[idx];
    }
    psum[t] = s;
    __syncthreads();
    if (t == 0) {
        unsigned run = 0;
        for (int i = 0; i < THREADS; ++i) { pbase[i] = run; run += psum[i]; }
    }
    __syncthreads();
    unsigned run = pbase[t];
    for (int i = 0; i < per; ++i) {
        int idx = t * per + i;
        if (idx < nchunks) { offsets[idx] = run; run += counts[idx]; }
    }
}

// ---------------------------------------------------------------------------
// Pass 3: fused 8-head MLP (WMMA f16 -> f32) + ReLU + w1 projection + ordered
// mask compaction scatter. 8 waves/block; each block processes
// CHUNKS_PER_BLOCK chunks of 128 rows so weight staging is amortized 8x.
// ---------------------------------------------------------------------------
__global__ __launch_bounds__(THREADS)
void SimplyInteraction_55637006352697_kernel(const float* __restrict__ x,
                                             const int* __restrict__ mask,
                                             const float* __restrict__ w0,
                                             const float* __restrict__ b0,
                                             const float* __restrict__ w1,
                                             const float* __restrict__ b1,
                                             const unsigned* __restrict__ chunk_base,
                                             float* __restrict__ out,
                                             int N, int nchunks, unsigned out_elems) {
    // w0 staged transposed [d][q][k] as f16 so B fragments are contiguous LDS reads
    __shared__ __align__(16) _Float16 w0t[INTDIM * HID * XDIM];  // 32 KB
    __shared__ float b0s[INTDIM * HID];
    __shared__ float w1s[INTDIM * HID];
    __shared__ float b1s[INTDIM];
    __shared__ float out_tile[ROWS_PER_CHUNK * INTDIM];          // [local_row][d]
    __shared__ unsigned excl[THREADS];

    const int t = threadIdx.x;
    const long long total = (long long)N * INTDIM;

    // Stage weights into LDS once per block (transpose w0 [d][k][q] -> [d][q][k]).
    for (int i = t; i < INTDIM * XDIM * HID; i += THREADS) {
        int d = i >> 11;            // / (XDIM*HID) == /2048
        int k = (i >> 6) & (XDIM - 1);
        int q = i & (HID - 1);
        w0t[(d * HID + q) * XDIM + k] = (_Float16)w0[i];
    }
    for (int i = t; i < INTDIM * HID; i += THREADS) {
        b0s[i] = b0[i];
        w1s[i] = w1[i];              // w1 is [d][HID][1] -> same flat layout
    }
    if (t < INTDIM) b1s[t] = b1[t];
    __syncthreads();

    const int lane   = t & 31;
    const int wave   = t >> 5;
    const int half16 = lane >> 4;    // which 16-lane half of the wave32
    const int l15    = lane & 15;
    const int kbase  = half16 * 8;   // K-group per ISA 16-bit A fragment layout

    for (int ci = 0; ci < CHUNKS_PER_BLOCK; ++ci) {
        const int cidx = blockIdx.x * CHUNKS_PER_BLOCK + ci;
        if (cidx >= nchunks) break;              // block-uniform exit
        const long long base_e = (long long)cidx * ELEMS_PER_CHUNK;
        const int row = cidx * ROWS_PER_CHUNK + wave * 16 + l15;

        // Warm the mask lines needed by the scatter phase of this chunk.
        if (base_e + (long long)t * 4 < total)
            __builtin_prefetch(mask + base_e + (long long)t * 4, 0, 1);

        // A fragment: lane l holds row (l&15); halves 0-7 = K kbase..kbase+7,
        // halves 8-15 = K kbase+16..kbase+23 (16-bit 16x32 A layout).
        v16h a;
        if (row < N) {
            const float* xr = x + (long long)row * XDIM + kbase;
            v4f x0 = *(const v4f*)(xr);
            v4f x1 = *(const v4f*)(xr + 4);
            v4f x2 = *(const v4f*)(xr + 16);
            v4f x3 = *(const v4f*)(xr + 20);
#pragma unroll
            for (int i = 0; i < 4; ++i) {
                a[i]      = (_Float16)x0[i];
                a[4 + i]  = (_Float16)x1[i];
                a[8 + i]  = (_Float16)x2[i];
                a[12 + i] = (_Float16)x3[i];
            }
        } else {
#pragma unroll
            for (int i = 0; i < 16; ++i) a[i] = (_Float16)0.0f;
        }

        // 8 heads x 4 column-tiles of the 64-wide hidden layer.
        for (int d = 0; d < INTDIM; ++d) {
            float acc[8] = {0.f, 0.f, 0.f, 0.f, 0.f, 0.f, 0.f, 0.f};
#pragma unroll
            for (int tt = 0; tt < 4; ++tt) {
                const int col = tt * 16 + l15;
                // B fragment: lane l holds column (l&15); same K grouping as A.
                const _Float16* bp = &w0t[(d * HID + col) * XDIM + kbase];
                union { v16h v; v8h h[2]; } bu;
                bu.h[0] = *(const v8h*)(bp);
                bu.h[1] = *(const v8h*)(bp + 16);
                v8f c = {};
                c = __builtin_amdgcn_wmma_f32_16x16x32_f16(
                        /*neg_a=*/false, a, /*neg_b=*/false, bu.v,
                        /*c_mod=*/(short)0, c, /*reuse_a=*/false, /*reuse_b=*/false);
                // Fuse bias + ReLU + layer-1 dot on the C fragment:
                // c[r] = h_pre[row = r + 8*half16][col]
                const float bias = b0s[d * HID + col];
                const float wv   = w1s[d * HID + col];
#pragma unroll
                for (int r = 0; r < 8; ++r) {
                    float hp = c[r] + bias;
                    acc[r] += (hp > 0.f ? hp : 0.f) * wv;
                }
            }
            // Reduce the 64 columns: sum across the 16 lanes of each half-wave.
#pragma unroll
            for (int r = 0; r < 8; ++r) {
                acc[r] += __shfl_xor(acc[r], 1, 32);
                acc[r] += __shfl_xor(acc[r], 2, 32);
                acc[r] += __shfl_xor(acc[r], 4, 32);
                acc[r] += __shfl_xor(acc[r], 8, 32);
            }
            if (l15 == 0) {
#pragma unroll
                for (int r = 0; r < 8; ++r) {
                    int lrow = wave * 16 + half16 * 8 + r;
                    out_tile[lrow * INTDIM + d] = acc[r] + b1s[d];
                }
            }
        }
        __syncthreads();

        // Ordered block-local compaction of this chunk's 1024 mask elements.
        int m[4];
        int c = 0;
#pragma unroll
        for (int i = 0; i < 4; ++i) {
            long long e = base_e + (long long)t * 4 + i;
            m[i] = (e < total && mask[e] != 0) ? 1 : 0;
            c += m[i];
        }
        excl[t] = (unsigned)c;
        __syncthreads();
        if (t == 0) {
            unsigned run = 0;
            for (int i = 0; i < THREADS; ++i) { unsigned v = excl[i]; excl[i] = run; run += v; }
        }
        __syncthreads();
        unsigned pos = chunk_base[cidx] + excl[t];
#pragma unroll
        for (int i = 0; i < 4; ++i) {
            if (m[i]) {
                int j = t * 4 + i;           // j = local_row * 8 + d
                if (pos < out_elems) out[pos] = out_tile[j];
                ++pos;
            }
        }
        __syncthreads();   // out_tile / excl reused next chunk
    }
}

extern "C" void kernel_launch(void* const* d_in, const int* in_sizes, int n_in,
                              void* d_out, int out_size, void* d_ws, size_t ws_size,
                              hipStream_t stream) {
    const float* x   = (const float*)d_in[0];   // [N, 32]
    const int*   msk = (const int*)d_in[1];     // [N, 8]
    const float* w0  = (const float*)d_in[2];   // [8, 32, 64]
    const float* b0  = (const float*)d_in[3];   // [8, 64]
    const float* w1  = (const float*)d_in[4];   // [8, 64, 1]
    const float* b1  = (const float*)d_in[5];   // [8, 1]
    float* out = (float*)d_out;

    const int N = in_sizes[0] / XDIM;
    const long long total = (long long)N * INTDIM;
    const int nchunks = (N + ROWS_PER_CHUNK - 1) / ROWS_PER_CHUNK;
    const int nblocks = (nchunks + CHUNKS_PER_BLOCK - 1) / CHUNKS_PER_BLOCK;

    unsigned* counts  = (unsigned*)d_ws;
    size_t off = ((size_t)nchunks * sizeof(unsigned) + 255) & ~(size_t)255;
    unsigned* offsets = (unsigned*)((char*)d_ws + off);

    si_count_kernel<<<nchunks, THREADS, 0, stream>>>(msk, counts, total);
    si_scan_kernel<<<1, THREADS, 0, stream>>>(counts, offsets, nchunks);
    SimplyInteraction_55637006352697_kernel<<<nblocks, THREADS, 0, stream>>>(
        x, msk, w0, b0, w1, b1, offsets, out, N, nchunks, (unsigned)out_size);
}
